// CADDY_69526930587955
// MI455X (gfx1250) — matrix operand order repaired and checked by reading, not compiled, not run on hardware
//
#include <hip/hip_runtime.h>
#include <hip/hip_bf16.h>
#include <math.h>

// MI455X / gfx1250, wave32. Fused temporal-graph-attention forward.
// WMMA f32<-f16 16x16x32 for the three matmul-shaped stages; double-buffered
// LDS slab staging for B operands; global_prefetch for the bank_feat stream.

typedef __attribute__((ext_vector_type(16))) _Float16 v16h;
typedef __attribute__((ext_vector_type(8)))  float    v8f;

#define K_N    32     // max neighbors
#define W_B    8      // bank window
#define D_E    128    // embed dim
#define H_O    256    // hidden / output dim
#define TWO_D  256
#define ROWS_PER_BLOCK 16

// A-fragment (16x32 f16, MxK) lane/element decomposition for a K-offset dr in [0,32):
//   element j of lane (hi*16 + M) holds K = hi*16 + j
__device__ __forceinline__ void aDecomp(int dr, int& hi, int& j) {
    hi = (dr >> 4) & 1;
    j  = dr & 15;
}

__device__ __forceinline__ v8f wmma16(v16h a, v16h b, v8f c) {
    // D = A(16x32 f16) * B(32x16 f16) + C(16x16 f32)
    return __builtin_amdgcn_wmma_f32_16x16x32_f16(
        /*neg_a=*/false, a, /*neg_b=*/false, b,
        /*c_mod=*/(short)0, c, /*reuse_a=*/false, /*reuse_b=*/false);
}

__global__ __launch_bounds__(256, 1)
void CADDY_tgn_fused_kernel(
    const float* __restrict__ dt_self,     // [N]
    const float* __restrict__ gc_self,     // [N]
    const float* __restrict__ dt_neigh,    // [N,K]
    const float* __restrict__ gc_neigh,    // [N,K]
    const unsigned char* __restrict__ neigh_mask, // [N,K] bool8
    const float* __restrict__ bank_feat,   // [N,W,2D]
    const float* __restrict__ bank_dt,     // [N,W]
    const unsigned char* __restrict__ bank_mask,  // [N,W] bool8
    const float* __restrict__ w_time,      // [D]
    const float* __restrict__ b_time,      // [D]
    const float* __restrict__ w_node,      // [D]
    const float* __restrict__ b_node,      // [D]
    const float* __restrict__ Wq,          // [D,D] (q = f @ Wq)
    const float* __restrict__ Wk,          // [D,D]
    const float* __restrict__ v_att,       // [D]
    const float* __restrict__ weight,      // [H,2D]
    float* __restrict__ out,               // [N,H]
    int nrows)
{
    // ---- LDS (~113KB) ----
    __shared__ v16h WkSw[4 * 8 * 32];       // 32KB: Wk fragments [ks][nt][lane]
    __shared__ v16h nodeASw[4 * 32];        // 4KB : node A-frags [ks][lane]
    __shared__ v16h neighASw[2 * 4 * 32];   // 8KB : neigh A-frags [mt][ks][lane]
    __shared__ v16h featASw[8 * 32];        // 8KB : feature A-frags [ks][lane]
    __shared__ v16h scratch[2 * 16 * 32];   // 32KB: kk(f32, low half) / dbl-buffered B slabs
    __shared__ float    nodeP[16 * 128];    // 8KB : normalized node feats f32
    __shared__ float    qv[16 * 128];       // 8KB : q vectors f32
    __shared__ _Float16 neighP[32 * 128];   // 8KB : normalized neigh feats f16
    __shared__ float sWt[128], sBt[128], sWn[128], sBn[128], sVa[128];
    __shared__ float norms16[16];
    __shared__ float norms32[32], sAtt[32], sWsc[32], sBw[8];
    __shared__ float sCnt;

    const int t    = threadIdx.x;
    const int lane = t & 31;
    const int wv   = t >> 5;                // 8 waves
    const int r0   = blockIdx.x * ROWS_PER_BLOCK;

    float*    kkF    = reinterpret_cast<float*>(scratch);     // first 16KB only
    _Float16* slabH  = reinterpret_cast<_Float16*>(scratch);  // both 16KB halves
    _Float16* WkH    = reinterpret_cast<_Float16*>(WkSw);
    _Float16* nodeAH = reinterpret_cast<_Float16*>(nodeASw);
    _Float16* neighAH= reinterpret_cast<_Float16*>(neighASw);
    _Float16* featAH = reinterpret_cast<_Float16*>(featASw);

    // ================= Phase 0: params + Wk fragment staging =================
    if (t < 128) {
        sWt[t] = w_time[t]; sBt[t] = b_time[t];
        sWn[t] = w_node[t]; sBn[t] = b_node[t];
        sVa[t] = v_att[t];
    }
    if (t < 16) norms16[t] = 0.f;
    for (int p = 0; p < 64; ++p) {          // 128*128 elems / 256 thr
        int idx = p * 256 + t;
        int k = idx >> 7, n = idx & 127;    // coalesced read of Wk[k][n]
        float v = Wk[idx];
        int ks = k >> 5, kr = k & 31;
        int hi2 = kr >> 4, j = kr & 15;     // B-frag: K = hi*16 + j
        int nt = n >> 4,  nl = n & 15;      // lane = hi*16 + N
        WkH[(((ks * 8 + nt) * 32) + hi2 * 16 + nl) * 16 + j] = (_Float16)v;
    }
    // warm up first row's bank_feat while we do the node-feature phase
    __builtin_prefetch(bank_feat + (size_t)r0 * W_B * TWO_D + t * 32, 0, 3);
    __syncthreads();

    // ================= Phase 1: node features (featurize + L2 norm) ==========
    for (int p = 0; p < 8; ++p) {           // 16*128 / 256
        int idx = p * 256 + t;
        int rr = idx >> 7, d = idx & 127;
        int row = r0 + rr;
        float f = fabsf(dt_self[row]) * sWt[d] + sBt[d]
                + gc_self[row] * sWn[d] + sBn[d];
        nodeP[rr * 128 + d] = f;
        atomicAdd(&norms16[rr], f * f);     // ds_add_f32
    }
    __syncthreads();
    for (int p = 0; p < 8; ++p) {
        int idx = p * 256 + t;
        int rr = idx >> 7, d = idx & 127;
        float inv = 1.0f / fmaxf(sqrtf(norms16[rr]), 1e-12f);
        float fn = nodeP[rr * 128 + d] * inv;
        nodeP[rr * 128 + d] = fn;
        int ks = d >> 5, dr = d & 31, hi, j;
        aDecomp(dr, hi, j);
        nodeAH[((ks * 32) + hi * 16 + rr) * 16 + j] = (_Float16)fn;
    }
    __syncthreads();

    // ================= Phase 2: q = nodeF @ Wq via WMMA (dbl-buffered B) =====
    {
        // Wq k-slab [32 x 128] -> 8 B-frag tiles = 256 v16h = 8KB per buffer
        auto stageWq = [&](int ks, int buf) {
            _Float16* dst = slabH + buf * (256 * 16);
            for (int p = 0; p < 16; ++p) {
                int idx = p * 256 + t;
                int kl = idx >> 7, n = idx & 127;
                float v = Wq[(ks * 32 + kl) * 128 + n];
                int hi2 = kl >> 4, j = kl & 15;
                int nt = n >> 4,  nl = n & 15;
                dst[((nt * 32) + hi2 * 16 + nl) * 16 + j] = (_Float16)v;
            }
        };
        v8f acc = {0.f,0.f,0.f,0.f,0.f,0.f,0.f,0.f};
        stageWq(0, 0);
        __syncthreads();
        for (int ks = 0; ks < 4; ++ks) {
            if (ks < 3) stageWq(ks + 1, (ks + 1) & 1);   // overlap with WMMA
            v16h a = nodeASw[ks * 32 + lane];
            v16h b = scratch[(ks & 1) * 256 + wv * 32 + lane]; // N-tile = wave id
            acc = wmma16(a, b, acc);
            __syncthreads();
        }
        int mb = (lane >> 4) * 8, n = wv * 16 + (lane & 15);
        #pragma unroll
        for (int v = 0; v < 8; ++v) qv[(mb + v) * 128 + n] = acc[v];
    }
    __syncthreads();

    // ================= Phase 3: per-row neighbor attention ===================
    for (int rr = 0; rr < ROWS_PER_BLOCK; ++rr) {
        const int row = r0 + rr;
        const float* dtn = dt_neigh + row * K_N;
        const float* gcn = gc_neigh + row * K_N;
        const unsigned char* nm = neigh_mask + row * K_N;

        // prefetch next row's bank_feat slab (8KB) into caches while we work
        if (row + 1 < nrows)
            __builtin_prefetch(bank_feat + (size_t)(row + 1) * W_B * TWO_D + t * 32, 0, 3);

        if (t < 32) { norms32[t] = 0.f; sAtt[t] = 0.f; }
        if (t == 32) sCnt = 0.f;
        if (t >= 224) {                         // 32 threads: bank decay weights
            int w = t - 224;
            float bm = bank_mask[row * W_B + w] ? 1.f : 0.f;
            sBw[w] = expf(-0.5f * bank_dt[row * W_B + w]) * bm;
        }
        __syncthreads();

        // 3a featurize neighbors -> kkF scratch (f32), accumulate sq-norms
        for (int p = 0; p < 16; ++p) {          // 32*128 / 256
            int idx = p * 256 + t;
            int k = idx >> 7, d = idx & 127;
            float f = fabsf(dtn[k]) * sWt[d] + sBt[d] + gcn[k] * sWn[d] + sBn[d];
            kkF[k * 128 + d] = f;
            atomicAdd(&norms32[k], f * f);
        }
        __syncthreads();
        // normalize -> neighP (f16) + swizzled A fragments
        for (int p = 0; p < 16; ++p) {
            int idx = p * 256 + t;
            int k = idx >> 7, d = idx & 127;
            float inv = 1.0f / fmaxf(sqrtf(norms32[k]), 1e-12f);
            _Float16 h = (_Float16)(kkF[k * 128 + d] * inv);
            neighP[k * 128 + d] = h;
            int mt = k >> 4, m = k & 15;
            int ks = d >> 5, dr = d & 31, hi, j;
            aDecomp(dr, hi, j);
            neighAH[(((mt * 4 + ks) * 32) + hi * 16 + m) * 16 + j] = h;
        }
        __syncthreads();

        // 3b kk = neighF @ Wk : M=32 (2 tiles) x N=128 (8 tiles, one per wave)
        {
            v8f acc0 = {0.f,0.f,0.f,0.f,0.f,0.f,0.f,0.f};
            v8f acc1 = {0.f,0.f,0.f,0.f,0.f,0.f,0.f,0.f};
            #pragma unroll
            for (int ks = 0; ks < 4; ++ks) {
                v16h b  = WkSw[(ks * 8 + wv) * 32 + lane];
                v16h a0 = neighASw[(0 * 4 + ks) * 32 + lane];
                v16h a1 = neighASw[(1 * 4 + ks) * 32 + lane];
                acc0 = wmma16(a0, b, acc0);
                acc1 = wmma16(a1, b, acc1);
            }
            int mb = (lane >> 4) * 8, n = wv * 16 + (lane & 15);
            #pragma unroll
            for (int v = 0; v < 8; ++v) {
                kkF[(mb + v) * 128 + n]        = acc0[v];   // k rows 0..15
                kkF[(16 + mb + v) * 128 + n]   = acc1[v];   // k rows 16..31
            }
        }
        __syncthreads();

        // 3c att[k] = sum_d tanh(q[d] + kk[k][d]) * v_att[d]
        {
            int k = t & 31, g = t >> 5;         // 8 d-groups of 16
            const float* qrow = qv + rr * 128;
            float s = 0.f;
            #pragma unroll
            for (int dd = g * 16; dd < g * 16 + 16; ++dd)
                s += tanhf(qrow[dd] + kkF[k * 128 + dd]) * sVa[dd];
            atomicAdd(&sAtt[k], s);
            if (t < 32) atomicAdd(&sCnt, nm[k] ? 1.f : 0.f);
        }
        __syncthreads();

        // 3d scores: leaky_relu(att + 2/(2+dt)) * mask / max(cnt,1)
        if (t < 32) {
            int k = t;
            float x = sAtt[k] + 2.0f / (2.0f + dtn[k]);
            float sc = x > 0.f ? x : 0.01f * x;
            float m = nm[k] ? 1.f : 0.f;
            sWsc[k] = sc * m / fmaxf(sCnt, 1.0f);
        }
        __syncthreads();

        // 3e/3f agg + bank history + write feature A-fragments (f16)
        if (t < 128) {
            int d = t;
            float agg = 0.f;
            #pragma unroll
            for (int k = 0; k < 32; ++k)
                agg += sWsc[k] * (float)neighP[k * 128 + d];
            const float* bf = bank_feat + (size_t)row * W_B * TWO_D;
            float b0 = 0.f, b1 = 0.f;
            #pragma unroll
            for (int w = 0; w < 8; ++w) {
                float bw = sBw[w];
                b0 += bw * bf[w * TWO_D + d];
                b1 += bw * bf[w * TWO_D + 128 + d];
            }
            float f0 = nodeP[rr * 128 + d] + b0;   // c = d      (node half)
            float f1 = agg + b1;                   // c = 128+d  (agg half)
            {
                int c = d, ks = c >> 5, dr = c & 31, hi, j;
                aDecomp(dr, hi, j);
                featAH[((ks * 32) + hi * 16 + rr) * 16 + j] = (_Float16)f0;
            }
            {
                int c = 128 + d, ks = c >> 5, dr = c & 31, hi, j;
                aDecomp(dr, hi, j);
                featAH[((ks * 32) + hi * 16 + rr) * 16 + j] = (_Float16)f1;
            }
        }
        __syncthreads();
    }

    // ================= Phase 4: out = relu(feat @ weight^T), dbl-buffered ====
    {
        // weight^T k-slab [32 x 256] -> 16 B-frag tiles = 512 v16h = 16KB/buffer
        auto stageW = [&](int ks, int buf) {
            _Float16* dst = slabH + buf * (512 * 16);
            for (int p = 0; p < 32; ++p) {      // 32*256 / 256
                int idx = p * 256 + t;
                int h = idx >> 5, dl = idx & 31;            // coalesced over dl
                float v = weight[h * TWO_D + ks * 32 + dl];
                int hi2 = dl >> 4, j = dl & 15;
                int nt = h >> 4,  nl = h & 15;
                dst[((nt * 32) + hi2 * 16 + nl) * 16 + j] = (_Float16)v;
            }
        };
        v8f acc0 = {0.f,0.f,0.f,0.f,0.f,0.f,0.f,0.f};
        v8f acc1 = {0.f,0.f,0.f,0.f,0.f,0.f,0.f,0.f};
        stageW(0, 0);
        __syncthreads();
        for (int ks = 0; ks < 8; ++ks) {
            if (ks < 7) stageW(ks + 1, (ks + 1) & 1);       // overlap with WMMA
            const v16h* buf = scratch + (ks & 1) * 512;
            v16h a  = featASw[ks * 32 + lane];
            v16h b0 = buf[(wv * 2 + 0) * 32 + lane];
            v16h b1 = buf[(wv * 2 + 1) * 32 + lane];
            acc0 = wmma16(a, b0, acc0);
            acc1 = wmma16(a, b1, acc1);
            __syncthreads();
        }
        int mb = (lane >> 4) * 8;
        int n0 = (wv * 2 + 0) * 16 + (lane & 15);
        int n1 = (wv * 2 + 1) * 16 + (lane & 15);
        #pragma unroll
        for (int v = 0; v < 8; ++v) {
            int row = r0 + mb + v;
            out[(size_t)row * H_O + n0] = fmaxf(acc0[v], 0.f);
            out[(size_t)row * H_O + n1] = fmaxf(acc1[v], 0.f);
        }
    }
}

extern "C" void kernel_launch(void* const* d_in, const int* in_sizes, int n_in,
                              void* d_out, int out_size, void* d_ws, size_t ws_size,
                              hipStream_t stream) {
    (void)n_in; (void)out_size; (void)d_ws; (void)ws_size;
    const float* dt_self    = (const float*)d_in[0];
    const float* gc_self    = (const float*)d_in[1];
    const float* dt_neigh   = (const float*)d_in[2];
    const float* gc_neigh   = (const float*)d_in[3];
    const unsigned char* neigh_mask = (const unsigned char*)d_in[4];
    const float* bank_feat  = (const float*)d_in[5];
    const float* bank_dt    = (const float*)d_in[6];
    const unsigned char* bank_mask  = (const unsigned char*)d_in[7];
    const float* w_time     = (const float*)d_in[8];
    const float* b_time     = (const float*)d_in[9];
    const float* w_node     = (const float*)d_in[10];
    const float* b_node     = (const float*)d_in[11];
    const float* Wq         = (const float*)d_in[12];
    const float* Wk         = (const float*)d_in[13];
    const float* v_att      = (const float*)d_in[14];
    const float* weight     = (const float*)d_in[15];
    float* out = (float*)d_out;

    int nrows  = in_sizes[0];                 // E*2 = 8192
    int blocks = nrows / ROWS_PER_BLOCK;      // 512 workgroups
    CADDY_tgn_fused_kernel<<<dim3(blocks), dim3(256), 0, stream>>>(
        dt_self, gc_self, dt_neigh, gc_neigh, neigh_mask,
        bank_feat, bank_dt, bank_mask,
        w_time, b_time, w_node, b_node, Wq, Wk, v_att, weight, out, nrows);
}
